// Attention_32882269618235
// MI455X (gfx1250) — matrix operand order repaired and checked by reading, not compiled, not run on hardware
//
#include <hip/hip_runtime.h>
#include <math.h>

// MI455X (gfx1250) fused attention forward, f16 WMMA matmuls + fp32 softmax.
// Outputs: out [B,H,S,D] followed by attn [B,H,S,S], concatenated flat in d_out.
//
// Pass 2 stages the shared V tile through LDS (f16, transposed) so the
// V^T WMMA A-fragments are built with 16B ds_load_b128 reads instead of
// 64 scattered global_load_b32 per lane; attn (1.07 GB stream) is written
// with non-temporal hints to keep the 192MB L2 for the reused K/V/mask.

typedef __attribute__((ext_vector_type(16))) _Float16 v16h;
typedef __attribute__((ext_vector_type(8)))  _Float16 v8h;
typedef __attribute__((ext_vector_type(8)))  float    v8f;
typedef __attribute__((ext_vector_type(4)))  float    f4;
typedef __attribute__((ext_vector_type(4)))  int      i4;

constexpr int BB = 4, HH = 16, SS = 2048, DD = 64;
constexpr int WPB = 8;                 // waves per block (256 threads)
constexpr float SCALE = 0.125f;        // 1/sqrt(64)
constexpr float MINF  = -3.402823466e38f;
constexpr int VSTRIDE = 40;            // LDS row stride (halves): 80B, 16B-aligned rows

__global__ __launch_bounds__(WPB * 32)
void attn_fwd(const float* __restrict__ Q, const float* __restrict__ K,
              const float* __restrict__ V, const int* __restrict__ Mk,
              float* __restrict__ Out, float* __restrict__ Attn)
{
  __shared__ _Float16 ldsV[DD * VSTRIDE];   // transposed V tile: [d][k], 5 KB

  const int lane = threadIdx.x & 31;
  const int wave = threadIdx.x >> 5;
  const int g    = lane >> 4;          // half-wave group (0/1)
  const int n    = lane & 15;          // C-layout column = q index in tile
  const int h = blockIdx.y, b = blockIdx.z;
  const int bh = b * HH + h;
  const int qbase = blockIdx.x * (16 * WPB) + wave * 16;
  const int qrow  = qbase + n;

  const float* Qb = Q + (size_t)bh * SS * DD;
  const float* Kb = K + (size_t)bh * SS * DD;
  const float* Vb = V + (size_t)bh * SS * DD;
  const int*   Mb = Mk + ((size_t)b * SS + qrow) * SS;
  float*       Ab = Attn + ((size_t)bh * SS + qrow) * SS;

  // ---- Q^T B-fragments (B 32x16: half j -> d = c*32 + 16g + j), kept live ----
  v16h bq[2];
#pragma unroll
  for (int c = 0; c < 2; ++c) {
    const float* qp = Qb + (size_t)qrow * DD + c * 32 + g * 16;
    f4 x0 = *(const f4*)(qp + 0);
    f4 x1 = *(const f4*)(qp + 4);
    f4 x2 = *(const f4*)(qp + 8);
    f4 x3 = *(const f4*)(qp + 12);
#pragma unroll
    for (int j = 0; j < 4; ++j) {
      bq[c][j]      = (_Float16)x0[j];
      bq[c][4 + j]  = (_Float16)x1[j];
      bq[c][8 + j]  = (_Float16)x2[j];
      bq[c][12 + j] = (_Float16)x3[j];
    }
  }

  // S^T tile for 16 keys starting at kb: A = K rows (M=key, Kdim=d), B = Q^T.
  // A 16-bit layout: half j -> d = base + 16g + j.
  auto score_tile = [&](int kb) -> v8f {
    v8f c = {};
#pragma unroll
    for (int cc = 0; cc < 2; ++cc) {
      const float* kp = Kb + (size_t)(kb + n) * DD + cc * 32 + g * 16;
      f4 y0 = *(const f4*)(kp + 0);
      f4 y1 = *(const f4*)(kp + 4);
      f4 y2 = *(const f4*)(kp + 8);
      f4 y3 = *(const f4*)(kp + 12);
      v16h a;
#pragma unroll
      for (int j = 0; j < 4; ++j) {
        a[j]      = (_Float16)y0[j];
        a[4 + j]  = (_Float16)y1[j];
        a[8 + j]  = (_Float16)y2[j];
        a[12 + j] = (_Float16)y3[j];
      }
      c = __builtin_amdgcn_wmma_f32_16x16x32_f16(false, a, false, bq[cc],
                                                 (short)0, c, false, false);
    }
    return c;
  };

  // ---- Pass 1: per-lane online softmax stats over this lane's k subset ----
  float m = MINF, l = 0.f;
  for (int kb = 0; kb < SS; kb += 16) {
    v8f c = score_tile(kb);
    const int* mp = Mb + kb + 8 * g;      // lane holds k = kb + 8g + r
    i4 mk0 = *(const i4*)(mp);
    i4 mk1 = *(const i4*)(mp + 4);
    float s[8];
#pragma unroll
    for (int r = 0; r < 8; ++r) {
      int mv = (r < 4) ? mk0[r] : mk1[r - 4];
      s[r] = mv ? c[r] * SCALE : MINF;
    }
    float tm = s[0];
#pragma unroll
    for (int r = 1; r < 8; ++r) tm = fmaxf(tm, s[r]);
    float mn = fmaxf(m, tm);
    float acc = 0.f;
#pragma unroll
    for (int r = 0; r < 8; ++r) acc += __expf(s[r] - mn);
    l = l * __expf(m - mn) + acc;
    m = mn;
  }
  // merge stats with partner half-wave (same q, other half of k space)
  {
    float m2 = __shfl_xor(m, 16);
    float l2 = __shfl_xor(l, 16);
    float mm = fmaxf(m, m2);
    l = l * __expf(m - mm) + l2 * __expf(m2 - mm);
    m = mm;
  }
  const float inv_l = (l > 0.f) ? 1.f / l : 0.f;

  // ---- Pass 2: recompute, write attn (NT), accumulate out^T = V^T * P^T ----
  v8f co[4] = {{}, {}, {}, {}};
  for (int kb = 0; kb < SS; kb += 32) {
    // Cooperatively stage V[kb..kb+31][0..63] into LDS as f16, transposed
    // to [d][k] so A-fragment reads are contiguous 16B chunks.
    {
      const int tid = (int)threadIdx.x;   // 0..255
      const int kk  = tid >> 3;           // 0..31
      const int d0  = (tid & 7) * 8;      // 0,8,...,56
      const float* vp = Vb + (size_t)(kb + kk) * DD + d0;
      f4 a0 = *(const f4*)(vp);
      f4 a1 = *(const f4*)(vp + 4);
      __syncthreads();                    // previous chunk's reads complete
#pragma unroll
      for (int j = 0; j < 4; ++j) {
        ldsV[(d0 + j)     * VSTRIDE + kk] = (_Float16)a0[j];
        ldsV[(d0 + 4 + j) * VSTRIDE + kk] = (_Float16)a1[j];
      }
      __syncthreads();                    // tile visible to all waves
    }

    float p0[8], p1[8];
#pragma unroll
    for (int t = 0; t < 2; ++t) {
      v8f c = score_tile(kb + 16 * t);
      const int* mp = Mb + kb + 16 * t + 8 * g;
      i4 mk0 = *(const i4*)(mp);
      i4 mk1 = *(const i4*)(mp + 4);
      float* p = t ? p1 : p0;
#pragma unroll
      for (int r = 0; r < 8; ++r) {
        int mv = (r < 4) ? mk0[r] : mk1[r - 4];
        p[r] = mv ? __expf(c[r] * SCALE - m) * inv_l : 0.f;  // masked -> exact 0
      }
      f4 w0, w1;
#pragma unroll
      for (int r = 0; r < 4; ++r) { w0[r] = p[r]; w1[r] = p[4 + r]; }
      f4* ap = (f4*)(Ab + kb + 16 * t + 8 * g);
      __builtin_nontemporal_store(w0, ap + 0);   // 1GB stream: keep out of L2
      __builtin_nontemporal_store(w1, ap + 1);
    }

    // Build P^T B-fragment (B 32x16: half j -> kk = 16g + j) via one
    // half-wave exchange: g0 needs partner's tile0, g1 needs partner's tile1.
    float recv[8];
#pragma unroll
    for (int r = 0; r < 8; ++r) {
      float send = g ? p0[r] : p1[r];
      recv[r] = __shfl_xor(send, 16);
    }
    v16h bp;
#pragma unroll
    for (int j = 0; j < 8; ++j) {
      bp[j]     = (_Float16)(g ? recv[j] : p0[j]);
      bp[8 + j] = (_Float16)(g ? p1[j]  : recv[j]);
    }

    // out^T[d,q] tiles: A = V^T (M=d, Kdim=kk), half j -> kk = 16g+j.
    // From transposed LDS tile: two contiguous 8-half (16B) runs per fragment.
#pragma unroll
    for (int dt = 0; dt < 4; ++dt) {
      const _Float16* lp = &ldsV[(dt * 16 + n) * VSTRIDE + 16 * g];
      v8h lo = *(const v8h*)(lp);        // kk = 16g .. 16g+7
      v8h hi = *(const v8h*)(lp + 8);    // kk = 16g+8 .. 16g+15
      v16h av;
#pragma unroll
      for (int j = 0; j < 8; ++j) { av[j] = lo[j]; av[8 + j] = hi[j]; }
      co[dt] = __builtin_amdgcn_wmma_f32_16x16x32_f16(false, av, false, bp,
                                                      (short)0, co[dt], false, false);
    }
  }

  // ---- store out: C layout row = d-in-tile (r+8g), col = q (n) ----
#pragma unroll
  for (int dt = 0; dt < 4; ++dt) {
    f4 w0, w1;
#pragma unroll
    for (int r = 0; r < 4; ++r) { w0[r] = co[dt][r]; w1[r] = co[dt][4 + r]; }
    f4* op = (f4*)(Out + ((size_t)bh * SS + qrow) * DD + dt * 16 + 8 * g);
    op[0] = w0;
    op[1] = w1;
  }
}

extern "C" void kernel_launch(void* const* d_in, const int* in_sizes, int n_in,
                              void* d_out, int out_size, void* d_ws, size_t ws_size,
                              hipStream_t stream) {
  (void)in_sizes; (void)n_in; (void)out_size; (void)d_ws; (void)ws_size;
  const float* Q  = (const float*)d_in[0];
  const float* K  = (const float*)d_in[1];
  const float* V  = (const float*)d_in[2];
  const int*   Mk = (const int*)d_in[3];
  float* out  = (float*)d_out;
  float* attn = out + (size_t)BB * HH * SS * DD;   // outputs concatenated flat

  dim3 grid(SS / (16 * WPB), HH, BB);   // (16, 16, 4) blocks
  dim3 block(WPB * 32);                 // 8 wave32s
  attn_fwd<<<grid, block, 0, stream>>>(Q, K, V, Mk, out, attn);
}
